// MoEModel_3865470566681
// MI455X (gfx1250) — compile-verified, matrix operand order checked by reading, not run on hardware
//
#include <hip/hip_runtime.h>

typedef __attribute__((ext_vector_type(16))) _Float16 v16h;
typedef __attribute__((ext_vector_type(8)))  float    v8f;

union Frag16 { v16h h; unsigned int u[8]; };

// ---------------------------------------------------------------------------
// Kernel 1: per-image fused conv1+relu+pool1 (VALU) -> conv2 implicit GEMM
// (v_wmma_f32_16x16x32_f16) + relu + pool2 -> f16 feature vector [1600]
// ---------------------------------------------------------------------------
__global__ __launch_bounds__(256) void moe_conv_fused(
    const float* __restrict__ x,   // [8192,28,28,1]
    const float* __restrict__ w1,  // [3,3,1,32]
    const float* __restrict__ b1,  // [32]
    const float* __restrict__ w2,  // [3,3,32,64]
    const float* __restrict__ b2,  // [64]
    _Float16* __restrict__ fbuf)   // [8192,1600] f16
{
    __shared__ float s_img[784];
    __shared__ float s_w1[288];
    __shared__ float s_b1[32];
    __shared__ float s_b2[64];
    __shared__ __align__(16) _Float16 s_wT[9 * 64 * 32];   // [tap][co][ci]
    __shared__ __align__(16) _Float16 s_p1[169 * 32];      // pool1 [13][13][32]
    __shared__ unsigned int s_pool[1600];                  // pooled conv2 (bits of relu'd f32)

    const int tid = threadIdx.x;
    const int b   = blockIdx.x;

    // ---- stage-in ----
    for (int i = tid; i < 784; i += 256) s_img[i] = x[(size_t)b * 784 + i];
    for (int i = tid; i < 288; i += 256) s_w1[i] = w1[i];
    if (tid < 32) s_b1[tid] = b1[tid];
    if (tid < 64) s_b2[tid] = b2[tid];
    for (int i = tid; i < 9 * 32 * 64; i += 256) {
        int t = i >> 11, rem = i & 2047, ci = rem >> 6, co = rem & 63;
        s_wT[(t * 64 + co) * 32 + ci] = (_Float16)w2[i];
    }
    for (int i = tid; i < 1600; i += 256) s_pool[i] = 0u;
    __syncthreads();

    // ---- phase 1: conv1 + relu + 2x2 maxpool -> s_p1 (f16) ----
    for (int w = tid; w < 169 * 32; w += 256) {
        int p = w >> 5, ch = w & 31;
        int py = p / 13, px = p % 13;
        float best = 0.0f;  // relu floor
        #pragma unroll
        for (int pos = 0; pos < 4; ++pos) {
            int cy = 2 * py + (pos >> 1), cx = 2 * px + (pos & 1);
            float acc = s_b1[ch];
            #pragma unroll
            for (int ky = 0; ky < 3; ++ky)
                #pragma unroll
                for (int kx = 0; kx < 3; ++kx)
                    acc = fmaf(s_img[(cy + ky) * 28 + (cx + kx)],
                               s_w1[(ky * 3 + kx) * 32 + ch], acc);
            best = fmaxf(best, acc);
        }
        s_p1[p * 32 + ch] = (_Float16)best;
    }
    __syncthreads();

    // ---- phase 2: conv2 as implicit GEMM via WMMA ----
    // M = 10x10 conv positions (only rows/cols feeding VALID pool), K = 9 taps * 32ch, N = 64
    const int lane = tid & 31;
    const int wave = tid >> 5;
    const int hi   = (lane >> 4) & 1;
    // 16-bit A/B fragment K mapping (pairs): lanes0-15: {0..7,16..23}, lanes16-31: +8
    int kp[8];
    #pragma unroll
    for (int j = 0; j < 8; ++j) kp[j] = ((j < 4) ? (2 * j) : (2 * j + 8)) + hi * 8;

    for (int tw = wave; tw < 28; tw += 8) {          // 7 m-tiles x 4 n-tiles
        const int mbase = (tw >> 2) * 16;
        const int nbase = (tw & 3) * 16;
        int posA = mbase + (lane & 15); if (posA > 99) posA = 99;  // clamp pad rows
        const int oy = posA / 10, ox = posA % 10;
        const int n  = nbase + (lane & 15);
        v8f acc = {};
        #pragma unroll
        for (int t = 0; t < 9; ++t) {                // one tap == one K-step of 32
            const int iy = oy + t / 3, ix = ox + t % 3;
            const unsigned int* ap = (const unsigned int*)&s_p1[(iy * 13 + ix) * 32];
            const unsigned int* bp = (const unsigned int*)&s_wT[(t * 64 + n) * 32];
            Frag16 A, Bf;
            #pragma unroll
            for (int j = 0; j < 8; ++j) { A.u[j] = ap[kp[j] >> 1]; Bf.u[j] = bp[kp[j] >> 1]; }
            acc = __builtin_amdgcn_wmma_f32_16x16x32_f16(false, A.h, false, Bf.h,
                                                         (short)0, acc, false, false);
        }
        // relu + 2x2 pool via monotone uint atomicMax (values >= 0)
        #pragma unroll
        for (int r = 0; r < 8; ++r) {
            int pos = mbase + r + hi * 8;
            if (pos < 100) {
                int ooy = pos / 10, oox = pos % 10;
                float v = fmaxf(acc[r] + s_b2[n], 0.0f);
                atomicMax(&s_pool[((ooy >> 1) * 5 + (oox >> 1)) * 64 + n], __float_as_uint(v));
            }
        }
    }
    __syncthreads();
    for (int i = tid; i < 1600; i += 256)
        fbuf[(size_t)b * 1600 + i] = (_Float16)__uint_as_float(s_pool[i]);
}

// ---------------------------------------------------------------------------
// Kernel 2: fused gate+experts GEMM [8192,1600]x[1600,55] on WMMA,
// then softmax-gate -> top-3 -> combine -> softmax.
// ---------------------------------------------------------------------------
__global__ __launch_bounds__(256) void moe_head(
    const _Float16* __restrict__ fbuf,      // [8192,1600] f16
    const float* __restrict__ gate_w,       // [1600,5]
    const float* __restrict__ gate_b,       // [5]
    const float* __restrict__ expert_w,     // [5,1600,10]
    const float* __restrict__ expert_b,     // [5,10]
    float* __restrict__ out)                // [8192,10]
{
    __shared__ __align__(16) _Float16 s_wp[64 * 32]; // packed weights, [col][k] per K-step
    __shared__ float s_out[128 * 64];                // per-block logits

    const int tid  = threadIdx.x;
    const int lane = tid & 31;
    const int wave = tid >> 5;
    const int hi   = (lane >> 4) & 1;
    int kp[8];
    #pragma unroll
    for (int j = 0; j < 8; ++j) kp[j] = ((j < 4) ? (2 * j) : (2 * j + 8)) + hi * 8;

    const long sbase = (long)blockIdx.x * 128 + wave * 16 + (lane & 15);

    v8f acc[4];
    #pragma unroll
    for (int nt = 0; nt < 4; ++nt) acc[nt] = (v8f){};

    for (int k0 = 0; k0 < 1600; k0 += 32) {
        // pack 55 weight columns (experts 0..49, gate 50..54, pad 55..63), transposed [col][k]
        for (int i = tid; i < 2048; i += 256) {
            int col = i & 63, kk = i >> 6, k = k0 + kk;
            float v = 0.0f;
            if (col < 50)      v = expert_w[((col / 10) * 1600 + k) * 10 + (col % 10)];
            else if (col < 55) v = gate_w[k * 5 + (col - 50)];
            s_wp[col * 32 + kk] = (_Float16)v;
        }
        __syncthreads();

        const unsigned int* ap = (const unsigned int*)(fbuf + sbase * 1600 + k0);
        Frag16 A;
        #pragma unroll
        for (int j = 0; j < 8; ++j) A.u[j] = ap[kp[j] >> 1];

        #pragma unroll
        for (int nt = 0; nt < 4; ++nt) {
            const int n = nt * 16 + (lane & 15);
            const unsigned int* bp = (const unsigned int*)&s_wp[n * 32];
            Frag16 Bf;
            #pragma unroll
            for (int j = 0; j < 8; ++j) Bf.u[j] = bp[kp[j] >> 1];
            acc[nt] = __builtin_amdgcn_wmma_f32_16x16x32_f16(false, A.h, false, Bf.h,
                                                             (short)0, acc[nt], false, false);
        }
        __syncthreads();
    }

    // spill logits to LDS for per-sample finalize
    #pragma unroll
    for (int nt = 0; nt < 4; ++nt)
        #pragma unroll
        for (int r = 0; r < 8; ++r)
            s_out[(wave * 16 + r + hi * 8) * 64 + nt * 16 + (lane & 15)] = acc[nt][r];
    __syncthreads();

    if (tid < 128) {
        const int ls = tid;
        const long s = (long)blockIdx.x * 128 + ls;
        // gate softmax over 5
        float g[5], gm = -1e30f;
        #pragma unroll
        for (int e = 0; e < 5; ++e) { g[e] = s_out[ls * 64 + 50 + e] + gate_b[e]; gm = fmaxf(gm, g[e]); }
        float gs = 0.0f;
        #pragma unroll
        for (int e = 0; e < 5; ++e) { g[e] = __expf(g[e] - gm); gs += g[e]; }
        float inv = 1.0f / gs;
        #pragma unroll
        for (int e = 0; e < 5; ++e) g[e] *= inv;
        // top-3 select + combine
        bool used[5] = {false, false, false, false, false};
        float comb[10];
        #pragma unroll
        for (int h = 0; h < 10; ++h) comb[h] = 0.0f;
        for (int kk = 0; kk < 3; ++kk) {
            int bi = 0; float bv = -1.0f;
            #pragma unroll
            for (int e = 0; e < 5; ++e)
                if (!used[e] && g[e] > bv) { bv = g[e]; bi = e; }
            used[bi] = true;
            #pragma unroll
            for (int h = 0; h < 10; ++h)
                comb[h] += bv * (s_out[ls * 64 + bi * 10 + h] + expert_b[bi * 10 + h]);
        }
        // final softmax over 10
        float cm = comb[0];
        #pragma unroll
        for (int h = 1; h < 10; ++h) cm = fmaxf(cm, comb[h]);
        float cs = 0.0f;
        #pragma unroll
        for (int h = 0; h < 10; ++h) { comb[h] = __expf(comb[h] - cm); cs += comb[h]; }
        inv = 1.0f / cs;
        #pragma unroll
        for (int h = 0; h < 10; ++h) out[s * 10 + h] = comb[h] * inv;
    }
}

extern "C" void kernel_launch(void* const* d_in, const int* in_sizes, int n_in,
                              void* d_out, int out_size, void* d_ws, size_t ws_size,
                              hipStream_t stream) {
    const float* x  = (const float*)d_in[0];
    const float* w1 = (const float*)d_in[1];
    const float* b1 = (const float*)d_in[2];
    const float* w2 = (const float*)d_in[3];
    const float* b2 = (const float*)d_in[4];
    const float* gw = (const float*)d_in[5];
    const float* gb = (const float*)d_in[6];
    const float* ew = (const float*)d_in[7];
    const float* eb = (const float*)d_in[8];

    _Float16* fbuf = (_Float16*)d_ws;  // 8192*1600 f16 = 26.2 MB scratch

    moe_conv_fused<<<8192, 256, 0, stream>>>(x, w1, b1, w2, b2, fbuf);
    moe_head<<<64, 256, 0, stream>>>(fbuf, gw, gb, ew, eb, (float*)d_out);
}
